// BotGAT_5531917877304
// MI455X (gfx1250) — matrix operand order repaired and checked by reading, not compiled, not run on hardware
//
#include <hip/hip_runtime.h>
#include <hip/hip_bf16.h>
#include <math.h>

// ---------------------------------------------------------------------------
// BotGAT forward for MI455X (gfx1250): bf16 WMMA GEMMs + atomic edge softmax
// ---------------------------------------------------------------------------

typedef __attribute__((ext_vector_type(16))) __bf16 v16bf;
typedef __attribute__((ext_vector_type(8)))  float  v8f;

__device__ __forceinline__ float leakyf(float x, float s) { return x >= 0.f ? x : s * x; }

// monotone float <-> uint encoding for atomicMax on floats
__device__ __forceinline__ unsigned fenc(float x) {
    unsigned u = __float_as_uint(x);
    return u ^ (unsigned)(((int)u >> 31) | 0x80000000);
}
__device__ __forceinline__ float fdec(unsigned u) {
    unsigned mask = (u >> 31) ? 0x80000000u : 0xFFFFFFFFu;
    return __uint_as_float(u ^ mask);
}

// ---------------------------------------------------------------------------
// Weight convert: W[K,M] fp32 (row-major) -> WT[M,Kpad] bf16, zero-padded in K
// ---------------------------------------------------------------------------
__global__ void conv_wT(const float* __restrict__ W, __bf16* __restrict__ WT,
                        int K, int M, int Kpad) {
    int t = blockIdx.x * blockDim.x + threadIdx.x;
    int total = M * Kpad;
    if (t >= total) return;
    int m = t / Kpad, k = t - m * Kpad;
    WT[t] = (k < K) ? (__bf16)W[(size_t)k * M + m] : (__bf16)0.f;
}

// ---------------------------------------------------------------------------
// GEMM: Y[:, colOff:colOff+16*MT] = act(X[nrows,Kact] @ W + bias)
//   X fp32 (leading dim ldx), W given as WT bf16 [M, Kpad] (pre-transposed),
//   fp32 accumulation via v_wmma_f32_16x16x32_bf16. One wave per 16-row tile,
//   4 waves / block, A tile staged through LDS (coalesced + f32->bf16 convert).
// ---------------------------------------------------------------------------
template <int MT>
__global__ __launch_bounds__(128) void gemm_wmma(
    const float* __restrict__ X, int ldx, int Kact,
    const __bf16* __restrict__ WT, int Kpad,
    const float* __restrict__ bias, float slope, int act,
    float* __restrict__ Y, int ldy, int colOff, int nrows)
{
    __shared__ __align__(32) __bf16 ldsA[4][16][32];
    const int lane = threadIdx.x & 31;
    const int wave = threadIdx.x >> 5;
    const int tile = blockIdx.x * 4 + wave;
    const int rowbase = tile * 16;
    const bool live = rowbase < nrows;

    // staging role: lane -> (row, 16-wide half of the 32-wide k tile)
    const int srow  = lane >> 1;
    const int shalf = (lane & 1) * 16;
    int grow = rowbase + srow;
    if (grow >= nrows) grow = nrows - 1;
    if (!live) grow = 0;
    const float* xrow = X + (size_t)grow * ldx;

    // fragment indexing (wave32 WMMA layout)
    const int hi   = lane >> 4;   // lane group 0/1
    const int mloc = lane & 15;

    v8f acc[MT];
#pragma unroll
    for (int c = 0; c < MT; ++c)
#pragma unroll
        for (int v = 0; v < 8; ++v) acc[c][v] = 0.f;

    for (int kb = 0; kb < Kpad; kb += 32) {
        __syncthreads();
        if (kb + 32 <= Kact) {
            // fast path (block-uniform): 16 consecutive floats -> bf16 -> LDS
            const float4* p = (const float4*)(xrow + kb + shalf);
            float4 f0 = p[0], f1 = p[1], f2 = p[2], f3 = p[3];
            v16bf t;
            t[0]  = (__bf16)f0.x; t[1]  = (__bf16)f0.y; t[2]  = (__bf16)f0.z; t[3]  = (__bf16)f0.w;
            t[4]  = (__bf16)f1.x; t[5]  = (__bf16)f1.y; t[6]  = (__bf16)f1.z; t[7]  = (__bf16)f1.w;
            t[8]  = (__bf16)f2.x; t[9]  = (__bf16)f2.y; t[10] = (__bf16)f2.z; t[11] = (__bf16)f2.w;
            t[12] = (__bf16)f3.x; t[13] = (__bf16)f3.y; t[14] = (__bf16)f3.z; t[15] = (__bf16)f3.w;
            *(v16bf*)&ldsA[wave][srow][shalf] = t;
        } else {
            // tail path: only for the K=6 / K=11 encoders (single k-block)
            v16bf t;
#pragma unroll
            for (int j = 0; j < 16; ++j) {
                int k = kb + shalf + j;
                t[j] = (k < Kact) ? (__bf16)xrow[k] : (__bf16)0.f;
            }
            *(v16bf*)&ldsA[wave][srow][shalf] = t;
        }
        if (kb + 32 < Kact) __builtin_prefetch(xrow + kb + 32 + shalf, 0, 1);
        __syncthreads();

        // A frag: lane m=mloc holds k = hi*16 + 0..15
        v16bf a;
#pragma unroll
        for (int j = 0; j < 8; ++j) {
            a[j]     = ldsA[wave][mloc][hi * 16 + j];
            a[j + 8] = ldsA[wave][mloc][hi * 16 + 8 + j];
        }
#pragma unroll
        for (int c = 0; c < MT; ++c) {
            // B frag: lane n=mloc holds k = kb + hi*16 + 0..15 -> contiguous 32B
            const __bf16* wp = WT + (size_t)(c * 16 + mloc) * Kpad + kb + hi * 16;
            v16bf b = *(const v16bf*)wp;
            acc[c] = __builtin_amdgcn_wmma_f32_16x16x32_bf16(
                false, a, false, b, (short)0, acc[c], false, false);
        }
    }
    if (!live) return;

    // C/D layout: lane n=mloc, VGPR v -> m = v + 8*hi
#pragma unroll
    for (int c = 0; c < MT; ++c) {
        int ncol = c * 16 + mloc;
        float bv = bias ? bias[ncol] : 0.f;
#pragma unroll
        for (int v = 0; v < 8; ++v) {
            int row = rowbase + v + hi * 8;
            if (row < nrows) {
                float r = acc[c][v] + bv;
                if (act) r = leakyf(r, slope);
                Y[(size_t)row * ldy + colOff + ncol] = r;
            }
        }
    }
}

// ---------------------------------------------------------------------------
// per-node attention scalars: a_s[n,h] = sum_c h[n,h*C+c]*att_s[h,c]
// ---------------------------------------------------------------------------
__global__ void attn_scal(const float* __restrict__ h,
                          const float* __restrict__ att_s, const float* __restrict__ att_d,
                          float* __restrict__ a_s, float* __restrict__ a_d,
                          int n, int H, int C) {
    int i = blockIdx.x * blockDim.x + threadIdx.x;
    if (i >= n) return;
    for (int hd = 0; hd < H; ++hd) {
        float ss = 0.f, sd = 0.f;
        const float4* hp = (const float4*)(h + (size_t)i * 128 + hd * C);
        const float4* as = (const float4*)(att_s + hd * C);
        const float4* ad = (const float4*)(att_d + hd * C);
        for (int c = 0; c < C / 4; ++c) {
            float4 v = hp[c], s4 = as[c], d4 = ad[c];
            ss += v.x * s4.x + v.y * s4.y + v.z * s4.z + v.w * s4.w;
            sd += v.x * d4.x + v.y * d4.y + v.z * d4.z + v.w * d4.w;
        }
        a_s[(size_t)i * H + hd] = ss;
        a_d[(size_t)i * H + hd] = sd;
    }
}

__global__ void init_ms(unsigned* __restrict__ m, float* __restrict__ s, int n) {
    int i = blockIdx.x * blockDim.x + threadIdx.x;
    if (i < n) { m[i] = 0u; s[i] = 0.f; }
}

__global__ void init_bias_out(float* __restrict__ out, const float* __restrict__ bias, size_t total) {
    size_t i = (size_t)blockIdx.x * blockDim.x + threadIdx.x;
    if (i < total) out[i] = bias[i & 127];
}

// ---------------------------------------------------------------------------
// edge pass 1: segment max of leaky(a_s[src]+a_d[dst], 0.2) via uint atomicMax
// ---------------------------------------------------------------------------
__global__ void edge_max(const int* __restrict__ ei, int E, int n,
                         const float* __restrict__ a_s, const float* __restrict__ a_d,
                         unsigned* __restrict__ m, int H) {
    int e = blockIdx.x * blockDim.x + threadIdx.x;
    int ET = E + n;
    if (e >= ET) return;
    int s_ = (e < E) ? ei[e] : (e - E);
    int d_ = (e < E) ? ei[E + e] : (e - E);
    for (int hd = 0; hd < H; ++hd) {
        float al = leakyf(a_s[(size_t)s_ * H + hd] + a_d[(size_t)d_ * H + hd], 0.2f);
        atomicMax(&m[(size_t)d_ * H + hd], fenc(al));
    }
}

// edge pass 2: segment sum of exp(alpha - max)
__global__ void edge_sum(const int* __restrict__ ei, int E, int n,
                         const float* __restrict__ a_s, const float* __restrict__ a_d,
                         const unsigned* __restrict__ m, float* __restrict__ ssum, int H) {
    int e = blockIdx.x * blockDim.x + threadIdx.x;
    int ET = E + n;
    if (e >= ET) return;
    int s_ = (e < E) ? ei[e] : (e - E);
    int d_ = (e < E) ? ei[E + e] : (e - E);
    for (int hd = 0; hd < H; ++hd) {
        float al = leakyf(a_s[(size_t)s_ * H + hd] + a_d[(size_t)d_ * H + hd], 0.2f);
        float ex = expf(al - fdec(m[(size_t)d_ * H + hd]));
        atomicAdd(&ssum[(size_t)d_ * H + hd], ex);
    }
}

// edge pass 3: out[dst] += h[src] * softmax-weight  (one wave32 per edge)
__global__ __launch_bounds__(256) void edge_agg(
    const int* __restrict__ ei, int E, int n,
    const float* __restrict__ a_s, const float* __restrict__ a_d,
    const unsigned* __restrict__ m, const float* __restrict__ ssum,
    const float* __restrict__ h, float* __restrict__ out, int H) {
    int lane = threadIdx.x & 31;
    int wid = blockIdx.x * (blockDim.x >> 5) + (threadIdx.x >> 5);
    int ET = E + n;
    if (wid >= ET) return;
    int s_ = (wid < E) ? ei[wid] : (wid - E);
    int d_ = (wid < E) ? ei[E + wid] : (wid - E);
    int C = 128 / H;
#pragma unroll
    for (int i = 0; i < 4; ++i) {
        int j = lane + 32 * i;
        int hd = j / C;
        float al = leakyf(a_s[(size_t)s_ * H + hd] + a_d[(size_t)d_ * H + hd], 0.2f);
        float w = expf(al - fdec(m[(size_t)d_ * H + hd])) /
                  (ssum[(size_t)d_ * H + hd] + 1e-16f);
        atomicAdd(&out[(size_t)d_ * 128 + j], h[(size_t)s_ * 128 + j] * w);
    }
}

// final tiny GEMM: out[n,2] = o1[n,128] @ w2[128,2] + b2
__global__ void final_out(const float* __restrict__ o1, const float* __restrict__ w2,
                          const float* __restrict__ b2, float* __restrict__ out, int n) {
    int i = blockIdx.x * blockDim.x + threadIdx.x;
    if (i >= n) return;
    float a0 = b2[0], a1 = b2[1];
    const float4* p = (const float4*)(o1 + (size_t)i * 128);
    for (int j = 0; j < 32; ++j) {
        float4 v = p[j];
        const float4* w01 = (const float4*)(w2 + 8 * j);       // cols interleaved
        float4 wa = w01[0], wb = w01[1];
        a0 += v.x * wa.x + v.y * wa.z + v.z * wb.x + v.w * wb.z;
        a1 += v.x * wa.y + v.y * wa.w + v.z * wb.y + v.w * wb.w;
    }
    out[2 * i] = a0;
    out[2 * i + 1] = a1;
}

// ---------------------------------------------------------------------------
// host orchestration
// ---------------------------------------------------------------------------
extern "C" void kernel_launch(void* const* d_in, const int* in_sizes, int n_in,
                              void* d_out, int out_size, void* d_ws, size_t ws_size,
                              hipStream_t stream) {
    const float* des      = (const float*)d_in[0];
    const float* tweet    = (const float*)d_in[1];
    const float* num_prop = (const float*)d_in[2];
    const float* cat_prop = (const float*)d_in[3];
    const int*   ei       = (const int*)d_in[4];
    // d_in[5] = edge_type (unused by reference)
    const float* w_des = (const float*)d_in[6],  *b_des = (const float*)d_in[7];
    const float* w_tw  = (const float*)d_in[8],  *b_tw  = (const float*)d_in[9];
    const float* w_np  = (const float*)d_in[10], *b_np  = (const float*)d_in[11];
    const float* w_cp  = (const float*)d_in[12], *b_cp  = (const float*)d_in[13];
    const float* w_in  = (const float*)d_in[14], *b_in  = (const float*)d_in[15];
    const float* g1_w  = (const float*)d_in[16], *g1_as = (const float*)d_in[17];
    const float* g1_ad = (const float*)d_in[18], *g1_b  = (const float*)d_in[19];
    const float* g2_w  = (const float*)d_in[20], *g2_as = (const float*)d_in[21];
    const float* g2_ad = (const float*)d_in[22], *g2_b  = (const float*)d_in[23];
    const float* w_o1  = (const float*)d_in[24], *b_o1  = (const float*)d_in[25];
    const float* w_o2  = (const float*)d_in[26], *b_o2  = (const float*)d_in[27];

    const int N = in_sizes[0] / 768;
    const int E = in_sizes[5];
    float* out = (float*)d_out;

    // workspace carve-up (256B aligned)
    char* ws = (char*)d_ws;
    size_t off = 0;
    auto take = [&](size_t bytes) { void* p = ws + off; off += (bytes + 255) & ~(size_t)255; return p; };
    float*    bufA  = (float*)take((size_t)N * 128 * 4);
    float*    bufB  = (float*)take((size_t)N * 128 * 4);
    float*    a_s   = (float*)take((size_t)N * 4 * 4);
    float*    a_d   = (float*)take((size_t)N * 4 * 4);
    unsigned* mMax  = (unsigned*)take((size_t)N * 4 * 4);
    float*    sSum  = (float*)take((size_t)N * 4 * 4);
    __bf16* wdesT = (__bf16*)take(32 * 768 * 2);
    __bf16* wtwT  = (__bf16*)take(32 * 768 * 2);
    __bf16* wnpT  = (__bf16*)take(32 * 32 * 2);
    __bf16* wcpT  = (__bf16*)take(32 * 32 * 2);
    __bf16* winT  = (__bf16*)take(128 * 128 * 2);
    __bf16* g1wT  = (__bf16*)take(128 * 128 * 2);
    __bf16* g2wT  = (__bf16*)take(128 * 128 * 2);
    __bf16* wo1T  = (__bf16*)take(128 * 128 * 2);
    (void)ws_size; (void)n_in; (void)out_size;

    auto cvt = [&](const float* W, __bf16* WT, int K, int M, int Kpad) {
        int total = M * Kpad;
        conv_wT<<<(total + 255) / 256, 256, 0, stream>>>(W, WT, K, M, Kpad);
    };
    cvt(w_des, wdesT, 768, 32, 768);
    cvt(w_tw,  wtwT,  768, 32, 768);
    cvt(w_np,  wnpT,  6,   32, 32);
    cvt(w_cp,  wcpT,  11,  32, 32);
    cvt(w_in,  winT,  128, 128, 128);
    cvt(g1_w,  g1wT,  128, 128, 128);
    cvt(g2_w,  g2wT,  128, 128, 128);
    cvt(w_o1,  wo1T,  128, 128, 128);

    const int ntiles = (N + 15) / 16;
    const int gblocks = (ntiles + 3) / 4;

    // encoders -> bufA[:, 0:128] (concat), leaky 0.01
    gemm_wmma<2><<<gblocks, 128, 0, stream>>>(des,      768, 768, wdesT, 768, b_des, 0.01f, 1, bufA, 128, 0,  N);
    gemm_wmma<2><<<gblocks, 128, 0, stream>>>(tweet,    768, 768, wtwT,  768, b_tw,  0.01f, 1, bufA, 128, 32, N);
    gemm_wmma<2><<<gblocks, 128, 0, stream>>>(num_prop, 6,   6,   wnpT,  32,  b_np,  0.01f, 1, bufA, 128, 64, N);
    gemm_wmma<2><<<gblocks, 128, 0, stream>>>(cat_prop, 11,  11,  wcpT,  32,  b_cp,  0.01f, 1, bufA, 128, 96, N);
    // x = leaky(x0 @ w_in + b_in) -> bufB
    gemm_wmma<8><<<gblocks, 128, 0, stream>>>(bufA, 128, 128, winT, 128, b_in, 0.01f, 1, bufB, 128, 0, N);

    const int ET = E + N;
    const int eb  = (ET + 255) / 256;
    const int ebw = (ET + 7) / 8;  // wave per edge, 8 waves/block

    // ---- GAT layer 1 (H=4, C=32) ----
    gemm_wmma<8><<<gblocks, 128, 0, stream>>>(bufB, 128, 128, g1wT, 128, nullptr, 0.f, 0, bufA, 128, 0, N);
    attn_scal<<<(N + 255) / 256, 256, 0, stream>>>(bufA, g1_as, g1_ad, a_s, a_d, N, 4, 32);
    init_ms<<<(N * 4 + 255) / 256, 256, 0, stream>>>(mMax, sSum, N * 4);
    init_bias_out<<<(int)(((size_t)N * 128 + 255) / 256), 256, 0, stream>>>(bufB, g1_b, (size_t)N * 128);
    edge_max<<<eb,  256, 0, stream>>>(ei, E, N, a_s, a_d, mMax, 4);
    edge_sum<<<eb,  256, 0, stream>>>(ei, E, N, a_s, a_d, mMax, sSum, 4);
    edge_agg<<<ebw, 256, 0, stream>>>(ei, E, N, a_s, a_d, mMax, sSum, bufA, bufB, 4);

    // ---- GAT layer 2 (H=1, C=128) ----
    gemm_wmma<8><<<gblocks, 128, 0, stream>>>(bufB, 128, 128, g2wT, 128, nullptr, 0.f, 0, bufA, 128, 0, N);
    attn_scal<<<(N + 255) / 256, 256, 0, stream>>>(bufA, g2_as, g2_ad, a_s, a_d, N, 1, 128);
    init_ms<<<(N + 255) / 256, 256, 0, stream>>>(mMax, sSum, N);
    init_bias_out<<<(int)(((size_t)N * 128 + 255) / 256), 256, 0, stream>>>(bufB, g2_b, (size_t)N * 128);
    edge_max<<<eb,  256, 0, stream>>>(ei, E, N, a_s, a_d, mMax, 1);
    edge_sum<<<eb,  256, 0, stream>>>(ei, E, N, a_s, a_d, mMax, sSum, 1);
    edge_agg<<<ebw, 256, 0, stream>>>(ei, E, N, a_s, a_d, mMax, sSum, bufA, bufB, 1);

    // ---- output MLP ----
    gemm_wmma<8><<<gblocks, 128, 0, stream>>>(bufB, 128, 128, wo1T, 128, b_o1, 0.01f, 1, bufA, 128, 0, N);
    final_out<<<(N + 255) / 256, 256, 0, stream>>>(bufA, w_o2, b_o2, out, N);
}
